// GraphGenerator_48180943127325
// MI455X (gfx1250) — matrix operand order repaired and checked by reading, not compiled, not run on hardware
//
#include <hip/hip_runtime.h>
#include <math.h>

// B=8, S=32, HN=64, TN=512, HID=EMB=512
typedef __attribute__((ext_vector_type(16))) __bf16 v16bf;
typedef __attribute__((ext_vector_type(8)))  float  v8f;
typedef __attribute__((ext_vector_type(4)))  unsigned int u32x4;

union BF16x16 {
  u32x4 u[2];
  v16bf v;
  unsigned short s[16];
};

__device__ __forceinline__ unsigned short f2bf(float f) {
  unsigned int u = __float_as_uint(f);
  unsigned int r = ((u >> 16) & 1u) + 0x7fffu;   // round-to-nearest-even
  return (unsigned short)((u + r) >> 16);
}
__device__ __forceinline__ float bf2f(unsigned short s) {
  return __uint_as_float(((unsigned int)s) << 16);
}

// ----------------------------- elementwise helpers -----------------------------

__global__ void conv_bf_kernel(const float* __restrict__ in, unsigned short* __restrict__ out, int n) {
  int i = blockIdx.x * 256 + threadIdx.x;
  if (i < n) out[i] = f2bf(in[i]);
}

// 512x512 f32 block -> bf16 transposed (out[n*512+k] = bf16(in[k*512+n]))
__global__ void convT_bf_kernel(const float* __restrict__ in, unsigned short* __restrict__ out) {
  __shared__ unsigned short tile[32][33];
  int k0 = blockIdx.x * 32, n0 = blockIdx.y * 32;
#pragma unroll
  for (int i = 0; i < 4; i++)
    tile[threadIdx.y + i * 8][threadIdx.x] =
        f2bf(in[(long)(k0 + threadIdx.y + i * 8) * 512 + n0 + threadIdx.x]);
  __syncthreads();
#pragma unroll
  for (int i = 0; i < 4; i++)
    out[(long)(n0 + threadIdx.y + i * 8) * 512 + k0 + threadIdx.x] =
        tile[threadIdx.x][threadIdx.y + i * 8];
}

__global__ void gather_bf_kernel(const float* __restrict__ table, const int* __restrict__ idx,
                                 unsigned short* __restrict__ out, int rows) {
  int i = blockIdx.x * 256 + threadIdx.x;
  if (i >= rows * 512) return;
  int r = i >> 9, c = i & 511;
  out[i] = f2bf(table[(long)idx[r] * 512 + c]);
}

// tq[b, s*64+hn, c] = elu(dterm2[b*32+s, c] + hterm[b*64+hn, c] + b_tq[c]), stored bf16
__global__ void build_tq_kernel(const float* __restrict__ dterm2, const float* __restrict__ hterm,
                                const float* __restrict__ b_tq, unsigned short* __restrict__ tq) {
  int i = blockIdx.x * 256 + threadIdx.x;           // 8*2048*512 = 8388608 exact
  int c = i & 511;
  int row = i >> 9;
  int b = row >> 11, r = row & 2047, s = r >> 6, hn = r & 63;
  float x = dterm2[(b * 32 + s) * 512 + c] + hterm[(b * 64 + hn) * 512 + c] + b_tq[c];
  x = x > 0.f ? x : expm1f(x);
  tq[i] = f2bf(x);
}

// tkT[b][h][t] = tk[b][t][h]  (bf16, LDS-tiled transpose)
__global__ void transpose_bf_kernel(const unsigned short* __restrict__ in,
                                    unsigned short* __restrict__ out) {
  __shared__ unsigned short tile[32][33];
  int b = blockIdx.z;
  int t0 = blockIdx.x * 32, h0 = blockIdx.y * 32;
  const unsigned short* ip = in + (long)b * 512 * 512;
  unsigned short* op = out + (long)b * 512 * 512;
#pragma unroll
  for (int i = 0; i < 4; i++)
    tile[threadIdx.y + i * 8][threadIdx.x] = ip[(t0 + threadIdx.y + i * 8) * 512 + h0 + threadIdx.x];
  __syncthreads();
#pragma unroll
  for (int i = 0; i < 4; i++)
    op[(h0 + threadIdx.y + i * 8) * 512 + t0 + threadIdx.x] = tile[threadIdx.x][threadIdx.y + i * 8];
}

// ----------------------------- WMMA bf16 GEMM (B pre-transposed) -----------------------------
// C[M,N] = act( A[M,K]@B (+ A2[M,K2]@B2) + extra[(row/extraDiv),:] + bias )
// A row-major bf16 (ld=K). BT is B transposed: BT[n*K + k] (ld=K). K multiple of 32, N mult of 64.
// Block: 128 threads = 4 waves; block tile 128x64; wave w owns rows w*32..w*32+31.
// No LDS, no barriers: A rows are wave-private, B reuse via L0/L2.
__global__ __launch_bounds__(128) void gemm_bf16_kernel(
    const unsigned short* __restrict__ A, const unsigned short* __restrict__ BT,
    const unsigned short* __restrict__ A2, const unsigned short* __restrict__ BT2,
    float* __restrict__ Cf, unsigned short* __restrict__ Cb,
    const float* __restrict__ bias, const float* __restrict__ extra, int extraDiv,
    int M, int N, int K, int K2, int act) {
  int tid = threadIdx.x;
  int lane = tid & 31, wv = tid >> 5;
  int rowBase = blockIdx.y * 128, colBase = blockIdx.x * 64;
  int r0 = wv * 32;
  int lrow = lane & 15;
  int koff = (lane >> 4) << 3;   // 0/8: A fragment half-offset
  int kb   = (lane >> 4) << 4;   // 0/16: added to brow, cancelled at B load (B uses koff)

  v8f acc[2][4] = {};

  for (int pass = 0; pass < 2; pass++) {
    const unsigned short* Ap = pass ? A2 : A;
    const unsigned short* Bp = pass ? BT2 : BT;
    int Kp = pass ? K2 : K;
    if (Ap == nullptr) continue;
    // clamp rows: loads stay in-bounds, results for row>=M are never stored
    int ga0 = min(rowBase + r0 + lrow, M - 1);
    int ga1 = min(rowBase + r0 + 16 + lrow, M - 1);
    const unsigned short* ar0 = Ap + (long)ga0 * Kp;
    const unsigned short* ar1 = Ap + (long)ga1 * Kp;
    const unsigned short* brow[4];
#pragma unroll
    for (int t = 0; t < 4; t++)
      brow[t] = Bp + (long)(colBase + t * 16 + lrow) * Kp + kb;

    for (int k0 = 0; k0 < Kp; k0 += 32) {
      BF16x16 a0, a1;
      a0.u[0] = *reinterpret_cast<const u32x4*>(ar0 + k0 + koff);
      a0.u[1] = *reinterpret_cast<const u32x4*>(ar0 + k0 + 16 + koff);
      a1.u[0] = *reinterpret_cast<const u32x4*>(ar1 + k0 + koff);
      a1.u[1] = *reinterpret_cast<const u32x4*>(ar1 + k0 + 16 + koff);
#pragma unroll
      for (int t = 0; t < 4; t++) {
        BF16x16 bfr;
        bfr.u[0] = *reinterpret_cast<const u32x4*>(brow[t] + k0 - kb + koff);
        bfr.u[1] = *reinterpret_cast<const u32x4*>(brow[t] + k0 - kb + 16 + koff);
        acc[0][t] = __builtin_amdgcn_wmma_f32_16x16x32_bf16(false, a0.v, false, bfr.v,
                                                            (short)0, acc[0][t], false, false);
        acc[1][t] = __builtin_amdgcn_wmma_f32_16x16x32_bf16(false, a1.v, false, bfr.v,
                                                            (short)0, acc[1][t], false, false);
      }
    }
  }
  // epilogue: C/D layout -> VGPR r: M=r (+8 for lanes>=16), N=lane&15
#pragma unroll
  for (int h = 0; h < 2; h++) {
#pragma unroll
    for (int t = 0; t < 4; t++) {
#pragma unroll
      for (int r = 0; r < 8; r++) {
        int row = rowBase + r0 + h * 16 + r + ((lane >> 4) << 3);
        int col = colBase + t * 16 + (lane & 15);
        if (row < M) {
          float x = acc[h][t][r];
          if (extra) x += extra[(long)(row / extraDiv) * N + col];
          if (bias) x += bias[col];
          if (act == 1) x = x > 0.f ? x : expm1f(x);
          else if (act == 2) x = tanhf(x);
          long o = (long)row * N + col;
          if (Cf) Cf[o] = x;
          if (Cb) Cb[o] = f2bf(x);
        }
      }
    }
  }
}

// ----------------------------- attention kernel -----------------------------
// One block per (b, 32 consecutive q rows == one s, half the heads).
// Phase1: scores (WMMA) -> LDS (masked); Phase2: exact softmax; Phase3: ctx = P@tk (WMMA).
__global__ __launch_bounds__(64) void attn_kernel(
    const unsigned short* __restrict__ tq, const unsigned short* __restrict__ tk,
    const unsigned short* __restrict__ tkT,
    const unsigned char* __restrict__ adj, const unsigned char* __restrict__ dup,
    unsigned short* __restrict__ tail_attn, unsigned short* __restrict__ ctx) {
  __shared__ float sc[32 * 512];   // 64 KB
  int tid = threadIdx.x, lane = tid & 31, wv = tid >> 5;
  int b = blockIdx.y;
  int q0 = blockIdx.x * 32;
  int s = q0 >> 6, hb = q0 & 63;
  const unsigned short* tqB  = tq  + ((long)(b * 2048 + q0)) * 512;
  const unsigned short* tkB  = tk  + (long)b * 512 * 512;    // [t][h]
  const unsigned short* tkTB = tkT + (long)b * 512 * 512;    // [h][t]
  int arow = wv * 16 + (lane & 15);
  int koff = (lane >> 4) << 3;
  int kb   = (lane >> 4) << 4;

  // Phase 1: scores[q, t] = sum_h tq[q,h]*tk[t,h]
  for (int c = 0; c < 8; c++) {
    v8f acc[4] = {};
    for (int h0 = 0; h0 < 512; h0 += 32) {
      BF16x16 afr;
      afr.u[0] = *reinterpret_cast<const u32x4*>(tqB + arow * 512 + h0 + koff);
      afr.u[1] = *reinterpret_cast<const u32x4*>(tqB + arow * 512 + h0 + 16 + koff);
#pragma unroll
      for (int t = 0; t < 4; t++) {
        int tt = c * 64 + t * 16 + (lane & 15);
        BF16x16 bfr;   // B col = key index tt, K = h -> contiguous tk row
        bfr.u[0] = *reinterpret_cast<const u32x4*>(tkB + (long)tt * 512 + h0 + koff);
        bfr.u[1] = *reinterpret_cast<const u32x4*>(tkB + (long)tt * 512 + h0 + 16 + koff);
        acc[t] = __builtin_amdgcn_wmma_f32_16x16x32_bf16(false, afr.v, false, bfr.v,
                                                         (short)0, acc[t], false, false);
      }
    }
#pragma unroll
    for (int t = 0; t < 4; t++) {
#pragma unroll
      for (int r = 0; r < 8; r++) {
        int qrow = wv * 16 + r + ((lane >> 4) << 3);
        int tt = c * 64 + t * 16 + (lane & 15);
        bool m = adj[((long)(b * 64 + hb + qrow)) * 512 + tt] &&
                 !dup[((long)(b * 32 + s)) * 512 + tt];
        sc[qrow * 512 + tt] = m ? acc[t][r] : -INFINITY;
      }
    }
  }
  __syncthreads();
  // Phase 2: exact softmax per row, write tail_attn (bf16)
  if (tid < 32) {
    float* rowp = &sc[tid * 512];
    float mx = -INFINITY;
    for (int t = 0; t < 512; t++) mx = fmaxf(mx, rowp[t]);
    float sum = 0.f;
    for (int t = 0; t < 512; t++) { float e = __expf(rowp[t] - mx); rowp[t] = e; sum += e; }
    float inv = 1.f / sum;
    unsigned short* tout = tail_attn + ((long)(b * 2048 + q0 + tid)) * 512;
    for (int t = 0; t < 512; t++) { float p = rowp[t] * inv; rowp[t] = p; tout[t] = f2bf(p); }
  }
  __syncthreads();
  // Phase 3: ctx[q, h] = sum_t P[q,t]*tk[t,h]  (B fragments from tkT, contiguous)
  for (int c = 0; c < 8; c++) {
    v8f acc[4] = {};
    for (int t0 = 0; t0 < 512; t0 += 32) {
      BF16x16 afr;
#pragma unroll
      for (int j = 0; j < 8; j++) afr.s[j]     = f2bf(sc[arow * 512 + t0 + koff + j]);
#pragma unroll
      for (int j = 0; j < 8; j++) afr.s[8 + j] = f2bf(sc[arow * 512 + t0 + 16 + koff + j]);
#pragma unroll
      for (int t = 0; t < 4; t++) {
        int hh = c * 64 + t * 16 + (lane & 15);
        BF16x16 bfr;
        bfr.u[0] = *reinterpret_cast<const u32x4*>(tkTB + (long)hh * 512 + t0 + koff);
        bfr.u[1] = *reinterpret_cast<const u32x4*>(tkTB + (long)hh * 512 + t0 + 16 + koff);
        acc[t] = __builtin_amdgcn_wmma_f32_16x16x32_bf16(false, afr.v, false, bfr.v,
                                                         (short)0, acc[t], false, false);
      }
    }
#pragma unroll
    for (int t = 0; t < 4; t++) {
#pragma unroll
      for (int r = 0; r < 8; r++) {
        int qrow = wv * 16 + r + ((lane >> 4) << 3);
        int col = c * 64 + t * 16 + (lane & 15);
        ctx[((long)(b * 2048 + q0 + qrow)) * 512 + col] = f2bf(acc[t][r]);
      }
    }
  }
}

// ----------------------------- final head attention + logprob -----------------------------
__global__ __launch_bounds__(128) void final_kernel(
    const float* __restrict__ hq, const unsigned short* __restrict__ hk,
    const unsigned short* __restrict__ tail_attn, const int* __restrict__ head,
    float* __restrict__ out) {
  int n = blockIdx.x;              // 0..255 = b*32+s
  int b = n >> 5, s = n & 31;
  __shared__ float sh[64];
  int tid = threadIdx.x;
  if (tid < 64) {
    float acc = 0.f;
    const float* q = hq + (long)n * 512;
    const unsigned short* kk = hk + ((long)(n * 64 + tid)) * 512;
    for (int j = 0; j < 512; j++) acc += q[j] * bf2f(kk[j]);
    int len = 0;
    for (int j = 0; j < 64; j++) len += (head[b * 64 + j] != 0);
    sh[tid] = (tid < len) ? acc : -INFINITY;
  }
  __syncthreads();
  if (tid == 0) {
    float mx = -INFINITY;
    for (int k = 0; k < 64; k++) mx = fmaxf(mx, sh[k]);
    float sum = 0.f;
    for (int k = 0; k < 64; k++) { float e = __expf(sh[k] - mx); sh[k] = e; sum += e; }
    float inv = 1.f / sum;
    for (int k = 0; k < 64; k++) sh[k] *= inv;
  }
  __syncthreads();
  const unsigned short* ta = tail_attn + ((long)(b * 2048 + s * 64)) * 512;
  for (int t = tid; t < 512; t += 128) {
    float p = 0.f;
    for (int k = 0; k < 64; k++) p += sh[k] * bf2f(ta[(long)k * 512 + t]);
    out[(long)n * 512 + t] = logf(p + 1e-20f);
  }
}

// ----------------------------- host launcher -----------------------------
extern "C" void kernel_launch(void* const* d_in, const int* in_sizes, int n_in,
                              void* d_out, int out_size, void* d_ws, size_t ws_size,
                              hipStream_t stream) {
  (void)in_sizes; (void)n_in; (void)out_size; (void)ws_size;
  const float* dec_out = (const float*)d_in[0];
  const float* history = (const float*)d_in[1];
  const int*   head    = (const int*)d_in[2];
  const int*   tail    = (const int*)d_in[3];
  const unsigned char* adj = (const unsigned char*)d_in[4];
  const unsigned char* dup = (const unsigned char*)d_in[5];
  const float* emb   = (const float*)d_in[6];
  const float* W_tq  = (const float*)d_in[7];
  const float* b_tq  = (const float*)d_in[8];
  const float* W_tk  = (const float*)d_in[9];
  const float* b_tk  = (const float*)d_in[10];
  const float* W_tout= (const float*)d_in[11];
  const float* W_hq  = (const float*)d_in[12];
  const float* b_hq  = (const float*)d_in[13];
  const float* W_hk  = (const float*)d_in[14];
  const float* b_hk  = (const float*)d_in[15];
  float* out = (float*)d_out;

  size_t off = 0;
  auto alloc = [&](size_t bytes) -> void* {
    void* p = (char*)d_ws + off;
    off += (bytes + 255) & ~(size_t)255;
    return p;
  };
  const size_t WSZ = (size_t)512 * 512 * 2;   // one transposed 512x512 bf16 weight block
  unsigned short* dec_bf     = (unsigned short*)alloc((size_t)256 * 512 * 2);
  unsigned short* hist_bf    = (unsigned short*)alloc((size_t)8 * 512 * 2);
  unsigned short* WdT        = (unsigned short*)alloc(WSZ);
  unsigned short* WhT        = (unsigned short*)alloc(WSZ);
  unsigned short* WeT        = (unsigned short*)alloc(WSZ);
  unsigned short* WtkT       = (unsigned short*)alloc(WSZ);
  unsigned short* Wt1T       = (unsigned short*)alloc(WSZ);
  unsigned short* Wt2T       = (unsigned short*)alloc(WSZ);
  unsigned short* Whq1T      = (unsigned short*)alloc(WSZ);
  unsigned short* Whq2T      = (unsigned short*)alloc(WSZ);
  unsigned short* WhkT       = (unsigned short*)alloc(WSZ);
  unsigned short* headEmb_bf = (unsigned short*)alloc((size_t)512 * 512 * 2);
  unsigned short* tailEmb_bf = (unsigned short*)alloc((size_t)4096 * 512 * 2);
  float* histterm            = (float*)alloc((size_t)8 * 512 * 4);
  float* E1                  = (float*)alloc((size_t)8 * 512 * 4);
  float* dterm2              = (float*)alloc((size_t)256 * 512 * 4);
  float* hterm               = (float*)alloc((size_t)512 * 512 * 4);
  unsigned short* tk_bf      = (unsigned short*)alloc((size_t)4096 * 512 * 2);
  unsigned short* tkTa       = (unsigned short*)alloc((size_t)4096 * 512 * 2);
  float* hq                  = (float*)alloc((size_t)256 * 512 * 4);
  unsigned short* tq_bf      = (unsigned short*)alloc((size_t)16384 * 512 * 2);
  unsigned short* tattn_bf   = (unsigned short*)alloc((size_t)16384 * 512 * 2);
  unsigned short* ctx_bf     = (unsigned short*)alloc((size_t)16384 * 512 * 2);
  unsigned short* ghid_bf    = (unsigned short*)alloc((size_t)16384 * 512 * 2);
  unsigned short* hk_bf      = (unsigned short*)alloc((size_t)16384 * 512 * 2);

  auto cdiv = [](int a, int b) { return (a + b - 1) / b; };

  conv_bf_kernel<<<cdiv(131072, 256), 256, 0, stream>>>(dec_out, dec_bf, 131072);
  conv_bf_kernel<<<cdiv(4096, 256),   256, 0, stream>>>(history, hist_bf, 4096);
  dim3 tg(16, 16), tb(32, 8);
  convT_bf_kernel<<<tg, tb, 0, stream>>>(W_tq,               WdT);
  convT_bf_kernel<<<tg, tb, 0, stream>>>(W_tq + 512 * 512,   WhT);
  convT_bf_kernel<<<tg, tb, 0, stream>>>(W_tq + 1024 * 512,  WeT);
  convT_bf_kernel<<<tg, tb, 0, stream>>>(W_tk,               WtkT);
  convT_bf_kernel<<<tg, tb, 0, stream>>>(W_tout,             Wt1T);
  convT_bf_kernel<<<tg, tb, 0, stream>>>(W_tout + 512 * 512, Wt2T);
  convT_bf_kernel<<<tg, tb, 0, stream>>>(W_hq,               Whq1T);
  convT_bf_kernel<<<tg, tb, 0, stream>>>(W_hq + 512 * 512,   Whq2T);
  convT_bf_kernel<<<tg, tb, 0, stream>>>(W_hk,               WhkT);
  gather_bf_kernel<<<cdiv(512 * 512, 256),  256, 0, stream>>>(emb, head, headEmb_bf, 512);
  gather_bf_kernel<<<cdiv(4096 * 512, 256), 256, 0, stream>>>(emb, tail, tailEmb_bf, 4096);

  dim3 blk(128);
  // histterm = history @ Wh
  gemm_bf16_kernel<<<dim3(8, 1), blk, 0, stream>>>(hist_bf, WhT, nullptr, nullptr,
      histterm, nullptr, nullptr, nullptr, 1, 8, 512, 512, 0, 0);
  // E1 = history @ W_hq[HID:]
  gemm_bf16_kernel<<<dim3(8, 1), blk, 0, stream>>>(hist_bf, Whq2T, nullptr, nullptr,
      E1, nullptr, nullptr, nullptr, 1, 8, 512, 512, 0, 0);
  // dterm2 = dec_out @ Wd + histterm[b]
  gemm_bf16_kernel<<<dim3(8, 2), blk, 0, stream>>>(dec_bf, WdT, nullptr, nullptr,
      dterm2, nullptr, nullptr, histterm, 32, 256, 512, 512, 0, 0);
  // hterm = head_emb @ We
  gemm_bf16_kernel<<<dim3(8, 4), blk, 0, stream>>>(headEmb_bf, WeT, nullptr, nullptr,
      hterm, nullptr, nullptr, nullptr, 1, 512, 512, 512, 0, 0);
  // tk = elu(tail_emb @ W_tk + b_tk)   (bf16)
  gemm_bf16_kernel<<<dim3(8, 32), blk, 0, stream>>>(tailEmb_bf, WtkT, nullptr, nullptr,
      nullptr, tk_bf, b_tk, nullptr, 1, 4096, 512, 512, 0, 1);
  // hq = elu(dec_out @ W_hq[:HID] + E1[b] + b_hq)  (f32)
  gemm_bf16_kernel<<<dim3(8, 2), blk, 0, stream>>>(dec_bf, Whq1T, nullptr, nullptr,
      hq, nullptr, b_hq, E1, 32, 256, 512, 512, 0, 1);

  build_tq_kernel<<<32768, 256, 0, stream>>>(dterm2, hterm, b_tq, tq_bf);
  transpose_bf_kernel<<<dim3(16, 16, 8), dim3(32, 8), 0, stream>>>(tk_bf, tkTa);

  attn_kernel<<<dim3(64, 8), 64, 0, stream>>>(tq_bf, tk_bf, tkTa, adj, dup, tattn_bf, ctx_bf);

  // graph_hid = tanh(ctx @ Wt1 + tq @ Wt2)  (fused dual-pass GEMM, bf16 out)
  gemm_bf16_kernel<<<dim3(8, 128), blk, 0, stream>>>(ctx_bf, Wt1T, tq_bf, Wt2T,
      nullptr, ghid_bf, nullptr, nullptr, 1, 16384, 512, 512, 512, 2);
  // hk = elu(graph_hid @ W_hk + b_hk)  (bf16)
  gemm_bf16_kernel<<<dim3(8, 128), blk, 0, stream>>>(ghid_bf, WhkT, nullptr, nullptr,
      nullptr, hk_bf, b_hk, nullptr, 1, 16384, 512, 512, 0, 1);

  final_kernel<<<256, 128, 0, stream>>>(hq, hk_bf, tattn_bf, head, out);
}
